// GNNEncoder_43817256354360
// MI455X (gfx1250) — matrix operand order repaired and checked by reading, not compiled, not run on hardware
//
#include <hip/hip_runtime.h>
#include <hip/hip_bf16.h>

typedef __attribute__((ext_vector_type(16))) __bf16 bf16x16;
typedef __attribute__((ext_vector_type(8)))  float  f32x8;

#define DIN 128           // feature dim entering each layer (128 for both layers)
#define NEG_SLOPE 0.2f

// ---------- order-preserving float<->uint for atomic segment-max ----------
__device__ __forceinline__ unsigned enc_ord(float f) {
    unsigned u = __float_as_uint(f);
    return (u & 0x80000000u) ? ~u : (u | 0x80000000u);
}
__device__ __forceinline__ float dec_ord(unsigned u) {
    return (u & 0x80000000u) ? __uint_as_float(u ^ 0x80000000u) : __uint_as_float(~u);
}
#define ENC_NEG_INF 0x007FFFFFu   // enc_ord(-inf)

// ---------- tiny precompute: wa[k] = sum_c W[k][c] * att[c] ----------
__global__ void k_watt(const float* __restrict__ Ws, const float* __restrict__ atts, float* __restrict__ was,
                       const float* __restrict__ Wd, const float* __restrict__ attd, float* __restrict__ wad,
                       int C) {
    int k = threadIdx.x;            // 128 threads
    float s = 0.f, d = 0.f;
    for (int c = 0; c < C; ++c) {
        s += Ws[k * C + c] * atts[c];
        d += Wd[k * C + c] * attd[c];
    }
    was[k] = s; wad[k] = d;
}

// ---------- transpose W[K=128][C] -> Wt[C][128] as bf16 hi/lo split ----------
__global__ void k_prep_w(const float* __restrict__ W, __bf16* __restrict__ th, __bf16* __restrict__ tl, int C) {
    int t = blockIdx.x * blockDim.x + threadIdx.x;
    if (t >= C * DIN) return;
    int c = t / DIN, k = t % DIN;
    float f = W[k * C + c];
    __bf16 h = (__bf16)f;
    th[c * DIN + k] = h;
    tl[c * DIN + k] = (__bf16)(f - (float)h);
}

// ---------- GEMM: H[n, C] = X[n, 128] @ W via bf16x3 split WMMA ----------
// Block = 256 threads = 8 waves. Block tile = 32 rows x C cols.
// Waves 0-3 -> row band 0 (16 rows), waves 4-7 -> band 1. Each wave owns C/64
// column tiles of 16, accumulating D = A x B + C with V_WMMA_F32_16X16X32_BF16.
template <int C>
__global__ __launch_bounds__(256) void k_gemm(const float* __restrict__ X,
                                              const __bf16* __restrict__ WTh,
                                              const __bf16* __restrict__ WTl,
                                              float* __restrict__ H, int nNodes) {
    constexpr int NT  = C / 16;   // column tiles
    constexpr int TPW = NT / 4;   // tiles per wave (4 waves per row band)
    const int wave = threadIdx.x >> 5;
    const int lane = threadIdx.x & 31;
    const int band = wave >> 2;
    const int wq   = wave & 3;
    const int half = lane >> 4;
    const int l15  = lane & 15;
    const int rowTile = blockIdx.x * 32 + band * 16;
    const int row = min(rowTile + l15, nNodes - 1);   // clamp: EXEC must stay all-ones for WMMA

    f32x8 acc[TPW];
    for (int t = 0; t < TPW; ++t)
        for (int r = 0; r < 8; ++r) acc[t][r] = 0.f;

    for (int ks = 0; ks < DIN / 32; ++ks) {
        const int kbase = ks * 32;
        // A fragment (16x32 bf16): lane&15 = M; lane-half selects the
        // K range [kbase+16*half .. +15] (matching B).
        // Consecutive halfwords = consecutive K.
        const float* xr = X + (size_t)row * DIN + kbase + half * 16;
        float av[16];
        const float4 f0 = *reinterpret_cast<const float4*>(xr);
        const float4 f1 = *reinterpret_cast<const float4*>(xr + 4);
        const float4 f2 = *reinterpret_cast<const float4*>(xr + 8);
        const float4 f3 = *reinterpret_cast<const float4*>(xr + 12);
        av[0]=f0.x; av[1]=f0.y; av[2]=f0.z; av[3]=f0.w;
        av[4]=f1.x; av[5]=f1.y; av[6]=f1.z; av[7]=f1.w;
        av[8]=f2.x; av[9]=f2.y; av[10]=f2.z; av[11]=f2.w;
        av[12]=f3.x; av[13]=f3.y; av[14]=f3.z; av[15]=f3.w;
        bf16x16 a_hi, a_lo;
        #pragma unroll
        for (int n = 0; n < 16; ++n) {
            __bf16 h = (__bf16)av[n];
            a_hi[n] = h;
            a_lo[n] = (__bf16)(av[n] - (float)h);
        }
        #pragma unroll
        for (int t = 0; t < TPW; ++t) {
            // B fragment (32x16 bf16): lane&15 = N; lane-half selects K range
            // [kbase+16*half .. +15], consecutive halfwords = consecutive K.
            const int ncol = (wq * TPW + t) * 16 + l15;
            const bf16x16 b_hi = *reinterpret_cast<const bf16x16*>(WTh + ncol * DIN + kbase + half * 16);
            const bf16x16 b_lo = *reinterpret_cast<const bf16x16*>(WTl + ncol * DIN + kbase + half * 16);
            // bf16x3 split product: hi*hi + hi*lo + lo*hi ~= fp32 product
            acc[t] = __builtin_amdgcn_wmma_f32_16x16x32_bf16(false, a_hi, false, b_hi, (short)0, acc[t], false, false);
            acc[t] = __builtin_amdgcn_wmma_f32_16x16x32_bf16(false, a_hi, false, b_lo, (short)0, acc[t], false, false);
            acc[t] = __builtin_amdgcn_wmma_f32_16x16x32_bf16(false, a_lo, false, b_hi, (short)0, acc[t], false, false);
        }
    }
    // D layout: lane&15 = N, M = r + 8*(lane>=16)
    #pragma unroll
    for (int t = 0; t < TPW; ++t) {
        const int ncol = (wq * TPW + t) * 16 + l15;
        #pragma unroll
        for (int r = 0; r < 8; ++r) {
            const int m = rowTile + r + 8 * half;
            if (m < nNodes) H[(size_t)m * C + ncol] = acc[t][r];
        }
    }
}

// ---------- per-node attention scalars: a[n] = x[n,:] . wa ----------
__global__ void k_row_dots(const float* __restrict__ X, const float* __restrict__ was,
                           const float* __restrict__ wad, float* __restrict__ as_,
                           float* __restrict__ ad_, int nNodes) {
    const int wave = threadIdx.x >> 5;
    const int lane = threadIdx.x & 31;
    const int node = blockIdx.x * 8 + wave;
    if (node >= nNodes) return;
    const float4 xv = *reinterpret_cast<const float4*>(X + (size_t)node * DIN + lane * 4);
    const float4 ws = *reinterpret_cast<const float4*>(was + lane * 4);
    const float4 wd = *reinterpret_cast<const float4*>(wad + lane * 4);
    float s = xv.x*ws.x + xv.y*ws.y + xv.z*ws.z + xv.w*ws.w;
    float d = xv.x*wd.x + xv.y*wd.y + xv.z*wd.z + xv.w*wd.w;
    for (int o = 16; o > 0; o >>= 1) {   // wave32 reduction
        s += __shfl_down(s, o);
        d += __shfl_down(d, o);
    }
    if (lane == 0) { as_[node] = s; ad_[node] = d; }
}

__global__ void k_init_seg(unsigned* __restrict__ segm, float* __restrict__ denom, int n) {
    int t = blockIdx.x * blockDim.x + threadIdx.x;
    if (t < n) { segm[t] = ENC_NEG_INF; denom[t] = 0.f; }
}

__global__ void k_zero(float* __restrict__ p, size_t n) {
    size_t t = (size_t)blockIdx.x * blockDim.x + threadIdx.x;
    if (t < n) p[t] = 0.f;
}

// ---------- edge phase ----------
__global__ void k_edge_logits(const int* __restrict__ src, const int* __restrict__ dst,
                              const float* __restrict__ as_, const float* __restrict__ ad_,
                              float* __restrict__ elog, unsigned* __restrict__ segm, int nE) {
    int e = blockIdx.x * blockDim.x + threadIdx.x;
    if (e >= nE) return;
    float l = as_[src[e]] + ad_[dst[e]];
    l = l > 0.f ? l : NEG_SLOPE * l;
    elog[e] = l;
    atomicMax(&segm[dst[e]], enc_ord(l));
}

__global__ void k_edge_exp(const int* __restrict__ dst, float* __restrict__ elog,
                           const unsigned* __restrict__ segm, float* __restrict__ denom, int nE) {
    int e = blockIdx.x * blockDim.x + threadIdx.x;
    if (e >= nE) return;
    const int d = dst[e];
    const float w = __expf(elog[e] - dec_ord(segm[d]));
    elog[e] = w;                       // in-place: logits buffer becomes exp-weights
    atomicAdd(&denom[d], w);
}

// out[dst] += alpha * h[src]; one wave covers C channels of 32/(C/4) edges
template <int C>
__global__ void k_edge_agg(const int* __restrict__ src, const int* __restrict__ dst,
                           const float* __restrict__ ew, const float* __restrict__ denom,
                           const float* __restrict__ H, float* __restrict__ AGG, int nE) {
    constexpr int TPE = C / 4;         // threads per edge, float4 each
    const int t  = blockIdx.x * blockDim.x + threadIdx.x;
    const int e  = t / TPE;
    const int c4 = t % TPE;
    if (e >= nE) return;
    const int s = src[e], d = dst[e];
    const float alpha = ew[e] / (denom[d] + 1e-16f);
    const float4 hv = *reinterpret_cast<const float4*>(H + (size_t)s * C + c4 * 4);
    float* o = AGG + (size_t)d * C + c4 * 4;
    atomicAdd(o + 0, alpha * hv.x);
    atomicAdd(o + 1, alpha * hv.y);
    atomicAdd(o + 2, alpha * hv.z);
    atomicAdd(o + 3, alpha * hv.w);
}

template <int C, bool RELU>
__global__ void k_bias_act(const float* __restrict__ agg, const float* __restrict__ b,
                           float* __restrict__ out, int nNodes) {
    size_t t = (size_t)blockIdx.x * blockDim.x + threadIdx.x;
    if (t >= (size_t)nNodes * C) return;
    float v = agg[t] + b[t % C];
    if (RELU) v = fmaxf(v, 0.f);
    out[t] = v;
}

extern "C" void kernel_launch(void* const* d_in, const int* in_sizes, int n_in,
                              void* d_out, int out_size, void* d_ws, size_t ws_size,
                              hipStream_t stream) {
    (void)n_in; (void)out_size; (void)ws_size;
    const float* x   = (const float*)d_in[0];
    const int*   ei  = (const int*)d_in[1];
    const float* W1s = (const float*)d_in[2];
    const float* W1d = (const float*)d_in[3];
    const float* a1s = (const float*)d_in[4];
    const float* a1d = (const float*)d_in[5];
    const float* b1  = (const float*)d_in[6];
    const float* W2s = (const float*)d_in[7];
    const float* W2d = (const float*)d_in[8];
    const float* a2s = (const float*)d_in[9];
    const float* a2d = (const float*)d_in[10];
    const float* b2  = (const float*)d_in[11];
    float* out = (float*)d_out;

    const int nN = in_sizes[0] / DIN;   // 100000
    const int nE = in_sizes[1] / 2;     // 1600000
    const int* src = ei;
    const int* dst = ei + nE;

    // workspace carve-out (256B aligned regions), ~136 MB total
    char* ws = (char*)d_ws;
    size_t off = 0;
    auto take = [&](size_t bytes) -> char* {
        char* p = ws + off;
        off += (bytes + 255) & ~(size_t)255;
        return p;
    };
    float*    hbuf   = (float*)take((size_t)nN * 128 * 4);  // h1, then act1 (layer-2 input)
    float*    aggbuf = (float*)take((size_t)nN * 128 * 4);  // per-layer aggregation
    float*    h2buf  = (float*)take((size_t)nN * 64 * 4);   // layer-2 messages
    float*    asrc   = (float*)take((size_t)nN * 4);
    float*    adst   = (float*)take((size_t)nN * 4);
    unsigned* segm   = (unsigned*)take((size_t)nN * 4);
    float*    denom  = (float*)take((size_t)nN * 4);
    float*    elog   = (float*)take((size_t)nE * 4);
    __bf16*   wth    = (__bf16*)take(128 * 128 * 2);
    __bf16*   wtl    = (__bf16*)take(128 * 128 * 2);
    float*    was    = (float*)take(128 * 4);
    float*    wad    = (float*)take(128 * 4);

    const dim3 B(256);
    const int gN  = (nN + 255) / 256;
    const int gE  = (nE + 255) / 256;
    const int gM  = (nN + 31) / 32;          // GEMM row blocks (3125)
    const int gRD = (nN + 7) / 8;            // row-dot waves

    // ===================== Layer 1 (128 -> 128, ReLU) =====================
    k_watt<<<1, 128, 0, stream>>>(W1s, a1s, was, W1d, a1d, wad, 128);
    k_prep_w<<<(128 * 128 + 255) / 256, B, 0, stream>>>(W1s, wth, wtl, 128);
    k_gemm<128><<<gM, B, 0, stream>>>(x, wth, wtl, hbuf, nN);
    k_row_dots<<<gRD, B, 0, stream>>>(x, was, wad, asrc, adst, nN);
    k_init_seg<<<gN, B, 0, stream>>>(segm, denom, nN);
    k_zero<<<(unsigned)(((size_t)nN * 128 + 255) / 256), B, 0, stream>>>(aggbuf, (size_t)nN * 128);
    k_edge_logits<<<gE, B, 0, stream>>>(src, dst, asrc, adst, elog, segm, nE);
    k_edge_exp<<<gE, B, 0, stream>>>(dst, elog, segm, denom, nE);
    k_edge_agg<128><<<(unsigned)(((size_t)nE * 32 + 255) / 256), B, 0, stream>>>(src, dst, elog, denom, hbuf, aggbuf, nE);
    k_bias_act<128, true><<<(unsigned)(((size_t)nN * 128 + 255) / 256), B, 0, stream>>>(aggbuf, b1, hbuf, nN);

    // ===================== Layer 2 (128 -> 64) =====================
    k_watt<<<1, 128, 0, stream>>>(W2s, a2s, was, W2d, a2d, wad, 64);
    k_prep_w<<<(128 * 64 + 255) / 256, B, 0, stream>>>(W2s, wth, wtl, 64);
    k_gemm<64><<<gM, B, 0, stream>>>(hbuf, wth, wtl, h2buf, nN);
    k_row_dots<<<gRD, B, 0, stream>>>(hbuf, was, wad, asrc, adst, nN);
    k_init_seg<<<gN, B, 0, stream>>>(segm, denom, nN);
    k_zero<<<(unsigned)(((size_t)nN * 64 + 255) / 256), B, 0, stream>>>(aggbuf, (size_t)nN * 64);
    k_edge_logits<<<gE, B, 0, stream>>>(src, dst, asrc, adst, elog, segm, nE);
    k_edge_exp<<<gE, B, 0, stream>>>(dst, elog, segm, denom, nE);
    k_edge_agg<64><<<(unsigned)(((size_t)nE * 16 + 255) / 256), B, 0, stream>>>(src, dst, elog, denom, h2buf, aggbuf, nE);
    k_bias_act<64, false><<<(unsigned)(((size_t)nN * 64 + 255) / 256), B, 0, stream>>>(aggbuf, b2, out, nN);
}
